// PointDiffusionUNet_84146999263769
// MI455X (gfx1250) — compile-verified, matrix operand order checked by reading, not run on hardware
//
#include <hip/hip_runtime.h>
#include <hip/hip_bf16.h>

// ---------------------------------------------------------------------------
// PointDiffusionUNet forward for MI455X (gfx1250, wave32, WMMA).
// Dense algebra: f32 operands are converted/transposed once into K-contiguous
// f16 mirrors, then a WMMA GEMM streams f16 tiles into LDS with double-
// buffered GLOBAL_LOAD_ASYNC_TO_LDS_B64 (+ s_wait_asynccnt) and feeds
// v_wmma_f32_16x16x32_f16 with plain ds_load_b128 fragments (zero converts).
// ---------------------------------------------------------------------------

#define USE_ASYNC_COPY 1

typedef __attribute__((ext_vector_type(16))) _Float16 v16h;
typedef __attribute__((ext_vector_type(8)))  _Float16 h8;
typedef __attribute__((ext_vector_type(8)))  float    v8f;

__device__ __forceinline__ float act_apply(float x, int act) {
  if (act == 1) return 0.5f * x * (1.0f + erff(x * 0.70710678118654752f)); // exact GELU
  if (act == 2) return x / (1.0f + __expf(-x));                           // SiLU
  return x;
}

#if USE_ASYNC_COPY
// 8-byte async copy: LDS[lds_off..+7] = g[0..7]   (tracked by ASYNCcnt)
__device__ __forceinline__ void async_ld_b64(unsigned lds_off, const void* g) {
  asm volatile("global_load_async_to_lds_b64 %0, %1, off"
               :: "v"(lds_off), "v"((unsigned long long)(size_t)g)
               : "memory");
}
__device__ __forceinline__ void async_wait0() {
  asm volatile("s_wait_asynccnt 0" ::: "memory");
}
#endif

// ---------------------------------------------------------------------------
// f32 -> f16 convert (+ implicit transpose): dst[b][r][k] = src[b][r*rs + k*cs]
// Produces K-contiguous rows for the GEMM's async tile staging.
// ---------------------------------------------------------------------------
__global__ void k_cvt(const float* __restrict__ src, long long rs, long long cs,
                      long long bs, _Float16* __restrict__ dst, int R, int Kd,
                      long long dbs)
{
  const long long b = blockIdx.y;
  const long long e = (long long)blockIdx.x * 256 + threadIdx.x;
  if (e >= (long long)R * Kd) return;
  const long long r = e / Kd, k = e % Kd;
  dst[b * dbs + e] = (_Float16)src[b * bs + r * rs + k * cs];
}

// ---------------------------------------------------------------------------
// WMMA GEMM on f16 mirrors:
//   Y[m,n] = act(alpha * sum_k A16[m][k]*B16[n][k] + bias[m]) + res[m,n]
// A16: [M][K] row-major halfs, B16: [N][K] (i.e. B^T) row-major halfs.
// Block = 128 threads (4 waves); tile = 16M x 64N x 32K; per-wave 16x16 C.
// LDS: A tile [16][32] + B tile [64][32] halfs, row stride 40 (bank-clean),
// double-buffered; fragments are two ds_load_b128 per operand, no converts.
// ---------------------------------------------------------------------------
__global__ __launch_bounds__(128)
void k_gemm(int M, int K, int N,
            const _Float16* __restrict__ A16, long long a_bs,
            const _Float16* __restrict__ B16, long long b_bs,
            const float* __restrict__ bias,
            const float* __restrict__ res, long long res_bs,
            float alpha, int act,
            float* __restrict__ Y, long long y_rs, long long y_cs, long long y_bs)
{
  __shared__ __align__(16) _Float16 lds[2 * 3200];  // 2 x (16*40 + 64*40) halfs
  const int ASTR = 40, BSTR = 40;                   // half-element strides

  const int b = blockIdx.z;
  A16 += (long long)b * a_bs;
  B16 += (long long)b * b_bs;
  Y   += (long long)b * y_bs;
  if (res) res += (long long)b * res_bs;

  const int t    = threadIdx.x;
  const int wave = t >> 5;
  const int lane = t & 31;
  const int m0   = blockIdx.y * 16;
  const int nblk = blockIdx.x * 64;

  // stage one K-tile (f16) into buffer p; clamped addresses -> branchless
  auto issue_tile = [&](int p, int k0) {
    _Float16* As = lds + p * 3200;
    _Float16* Bs = As + 16 * ASTR;
    {                                         // A: 16x32 halfs, 1 x b64/lane
      const int m = t >> 3, k4 = (t & 7) << 2;
      long long mm = m0 + m;  if (mm > M - 1) mm = M - 1;
      long long kk = k0 + k4; if (kk > K - 4) kk = K - 4;
      const _Float16* g = A16 + mm * K + kk;
#if USE_ASYNC_COPY
      async_ld_b64((unsigned)(size_t)&As[m * ASTR + k4], g);
#else
      *(float2*)&As[m * ASTR + k4] = *(const float2*)g;
#endif
    }
#pragma unroll
    for (int j = 0; j < 4; ++j) {             // B^T: 64x32 halfs, 4 x b64/lane
      const int e = t + 128 * j;
      const int n = e >> 3, k4 = (e & 7) << 2;
      long long nn = nblk + n; if (nn > N - 1) nn = N - 1;
      long long kk = k0 + k4;  if (kk > K - 4) kk = K - 4;
      const _Float16* g = B16 + nn * K + kk;
#if USE_ASYNC_COPY
      async_ld_b64((unsigned)(size_t)&Bs[n * BSTR + k4], g);
#else
      *(float2*)&Bs[n * BSTR + k4] = *(const float2*)g;
#endif
    }
  };

  issue_tile(0, 0);
  const int nk = (K + 31) >> 5;
  v8f acc = {};

  for (int it = 0; it < nk; ++it) {
    const int k0 = it << 5;
#if USE_ASYNC_COPY
    async_wait0();                 // this wave's copies for tile `it` landed
#endif
    __syncthreads();               // everyone's copies visible; prev reads done

    _Float16* As = lds + (it & 1) * 3200;
    _Float16* Bs = As + 16 * ASTR;

    if (k0 + 32 > K) {             // uniform K-tail: zero out-of-range columns
      const int kt = K - k0;
      for (int e = t; e < 512; e += 128) {
        const int m = e >> 5, k = e & 31;
        if (k >= kt) As[m * ASTR + k] = (_Float16)0.f;
      }
      for (int e = t; e < 2048; e += 128) {
        const int n = e >> 5, k = e & 31;
        if (k >= kt) Bs[n * BSTR + k] = (_Float16)0.f;
      }
      __syncthreads();
    }

    if (it + 1 < nk) issue_tile((it + 1) & 1, k0 + 32);  // overlap with compute

    // fragments (ISA 7.12.2 16-bit layout: K = (i/8)*16 + hi*8 + (i%8))
    // -> per lane: two contiguous 8-half runs = two ds_load_b128, no cvt.
    const _Float16* arow = As + (lane & 15) * ASTR;
    const _Float16* brow = Bs + (wave * 16 + (lane & 15)) * BSTR;
    const int hi8 = (lane >> 4) * 8;
    const h8 alo = *(const h8*)(arow + hi8);
    const h8 ahi = *(const h8*)(arow + 16 + hi8);
    const h8 blo = *(const h8*)(brow + hi8);
    const h8 bhi = *(const h8*)(brow + 16 + hi8);
    const v16h af = __builtin_shufflevector(alo, ahi, 0,1,2,3,4,5,6,7,
                                            8,9,10,11,12,13,14,15);
    const v16h bf = __builtin_shufflevector(blo, bhi, 0,1,2,3,4,5,6,7,
                                            8,9,10,11,12,13,14,15);
    acc = __builtin_amdgcn_wmma_f32_16x16x32_f16(false, af, false, bf,
                                                 (short)0, acc, false, false);
  }

  // C/D layout: element r -> (M = m0 + r + 8*hi, N = n0 + (lane&15))
  const int nn = nblk + wave * 16 + (lane & 15);
  const int hi = lane >> 4;
  if (nn < N) {
#pragma unroll
    for (int r = 0; r < 8; ++r) {
      const int mm = m0 + r + 8 * hi;
      if (mm < M) {
        float v = acc[r] * alpha;
        if (bias) v += bias[mm];
        v = act_apply(v, act);
        const long long o = (long long)mm * y_rs + (long long)nn * y_cs;
        if (res) v += res[o];
        Y[o] = v;
      }
    }
  }
}

// ---------------- LayerNorm over channels (x stored [C,N], LN over C) ------
__global__ __launch_bounds__(256)
void k_ln(const float* __restrict__ X, float* __restrict__ Y,
          const float* __restrict__ g, const float* __restrict__ bt,
          int C, int N, long long bs)
{
  const long long b = blockIdx.y;
  const int n = blockIdx.x * 256 + threadIdx.x;
  if (n >= N) return;
  const float* x = X + b * bs;
  float* y = Y + b * bs;
  float m = 0.f;
  for (int c = 0; c < C; ++c) m += x[(long long)c * N + n];
  m /= (float)C;
  float v = 0.f;
  for (int c = 0; c < C; ++c) { float d = x[(long long)c * N + n] - m; v += d * d; }
  v /= (float)C;
  const float inv = rsqrtf(v + 1e-5f);
  for (int c = 0; c < C; ++c)
    y[(long long)c * N + n] = (x[(long long)c * N + n] - m) * inv * g[c] + bt[c];
}

// ---------------- GroupNorm (+ optional activation), one block per (b,g) ---
__global__ __launch_bounds__(256)
void k_gn(const float* __restrict__ X, float* __restrict__ Y,
          const float* __restrict__ g, const float* __restrict__ bt,
          int C, int N, int G, int act, long long bs)
{
  __shared__ float s1[256], s2[256];
  const int b = blockIdx.y, grp = blockIdx.x;
  const float* x = X + (long long)b * bs;
  float* y = Y + (long long)b * bs;
  const int cg = C / G;
  const long long cnt = (long long)cg * N;
  const long long base = (long long)grp * cnt;
  float sm = 0.f, sq = 0.f;
  for (long long e = threadIdx.x; e < cnt; e += 256) {
    const float v = x[base + e]; sm += v; sq += v * v;
  }
  s1[threadIdx.x] = sm; s2[threadIdx.x] = sq; __syncthreads();
  for (int st = 128; st > 0; st >>= 1) {
    if ((int)threadIdx.x < st) { s1[threadIdx.x] += s1[threadIdx.x + st];
                                 s2[threadIdx.x] += s2[threadIdx.x + st]; }
    __syncthreads();
  }
  const float mean = s1[0] / (float)cnt;
  const float var  = s2[0] / (float)cnt - mean * mean;
  const float inv  = rsqrtf(var + 1e-5f);
  for (long long e = threadIdx.x; e < cnt; e += 256) {
    const int c = grp * cg + (int)(e / N);
    float v = (x[base + e] - mean) * inv * g[c] + bt[c];
    y[base + e] = act_apply(v, act);
  }
}

// ---------------- row softmax (in place), one block per row ----------------
__global__ __launch_bounds__(256)
void k_softmax(float* __restrict__ S, int L)
{
  __shared__ float red[256];
  float* row = S + (long long)blockIdx.x * L;
  const int tid = threadIdx.x;
  float mx = -3.4e38f;
  for (int i = tid; i < L; i += 256) mx = fmaxf(mx, row[i]);
  red[tid] = mx; __syncthreads();
  for (int st = 128; st > 0; st >>= 1) {
    if (tid < st) red[tid] = fmaxf(red[tid], red[tid + st]);
    __syncthreads();
  }
  mx = red[0]; __syncthreads();
  float sm = 0.f;
  for (int i = tid; i < L; i += 256) { float e = __expf(row[i] - mx); row[i] = e; sm += e; }
  red[tid] = sm; __syncthreads();
  for (int st = 128; st > 0; st >>= 1) {
    if (tid < st) red[tid] += red[tid + st];
    __syncthreads();
  }
  const float inv = 1.f / red[0];
  for (int i = tid; i < L; i += 256) row[i] *= inv;
}

// ---------------- farthest point sampling: one block per batch -------------
__global__ __launch_bounds__(256)
void k_fps(const float* __restrict__ xyz, long long xbs, int Np, int ns,
           int* __restrict__ idx, long long ibs)
{
  __shared__ float dist[8192];
  __shared__ float rv[256];
  __shared__ int   ri[256];
  __shared__ int   s_far;
  const float* P = xyz + (long long)blockIdx.x * xbs;
  int* out = idx + (long long)blockIdx.x * ibs;
  const int tid = threadIdx.x;
  for (int i = tid; i < Np; i += 256) dist[i] = 1e10f;
  if (tid == 0) s_far = 0;
  __syncthreads();
  for (int s = 0; s < ns; ++s) {
    const int far = s_far;
    if (tid == 0) out[s] = far;
    const float fx = P[far], fy = P[Np + far], fz = P[2 * Np + far];
    float bv = -1.f; int bi = 0;
    for (int i = tid; i < Np; i += 256) {
      const float dx = P[i] - fx, dy = P[Np + i] - fy, dz = P[2 * Np + i] - fz;
      const float d = dx * dx + dy * dy + dz * dz;
      const float nd = fminf(dist[i], d);
      dist[i] = nd;
      if (nd > bv) { bv = nd; bi = i; }
    }
    rv[tid] = bv; ri[tid] = bi;
    __syncthreads();
    for (int st = 128; st > 0; st >>= 1) {
      if (tid < st) {
        const float ov = rv[tid + st]; const int oi = ri[tid + st];
        if (ov > rv[tid] || (ov == rv[tid] && oi < ri[tid])) { rv[tid] = ov; ri[tid] = oi; }
      }
      __syncthreads();
    }
    if (tid == 0) s_far = ri[0];
    __syncthreads();
  }
}

// ---------------- gather along points ---------------------------------------
__global__ void k_gather(const float* __restrict__ src, long long sbs, int C, int Ns,
                         const int* __restrict__ idx, long long ibs, int M,
                         float* __restrict__ dst, long long dbs)
{
  const long long b = blockIdx.y;
  const long long e = (long long)blockIdx.x * 256 + threadIdx.x;
  if (e >= (long long)C * M) return;
  const int c = (int)(e / M), i = (int)(e % M);
  dst[b * dbs + (long long)c * M + i] =
      src[b * sbs + (long long)c * Ns + idx[b * ibs + i]];
}

// ---------------- nearest upsample along points ------------------------------
__global__ void k_up(const float* __restrict__ src, long long sbs, int C, int Ns,
                     float* __restrict__ dst, long long dbs, int Nt)
{
  const long long b = blockIdx.y;
  const long long e = (long long)blockIdx.x * 256 + threadIdx.x;
  if (e >= (long long)C * Nt) return;
  const int c = (int)(e / Nt), t = (int)(e % Nt);
  const int si = (int)((long long)t * Ns / Nt);
  dst[b * dbs + e] = src[b * sbs + (long long)c * Ns + si];
}

// ---------------- copy into channel slice (concat) ---------------------------
__global__ void k_copych(const float* __restrict__ src, long long sbs, int Cs, int N,
                         float* __restrict__ dst, long long dbs, int co)
{
  const long long b = blockIdx.y;
  const long long e = (long long)blockIdx.x * 256 + threadIdx.x;
  if (e >= (long long)Cs * N) return;
  dst[b * dbs + (long long)co * N + e] = src[b * sbs + e];
}

// ---------------- broadcast add of per-(b,c) vector --------------------------
__global__ void k_addvec(float* __restrict__ Y, const float* __restrict__ V,
                         int C, long long N, long long bs)
{
  const long long b = blockIdx.y;
  const long long e = (long long)blockIdx.x * 256 + threadIdx.x;
  if (e >= (long long)C * N) return;
  Y[b * bs + e] += V[b * C + e / N];
}

// ---------------- gelu(time embedding) --------------------------------------
__global__ void k_temb(const int* __restrict__ t, float* __restrict__ tg)
{
  const int b = blockIdx.x, j = threadIdx.x;  // 128 threads
  const float tv = (float)t[b];
  const int half = 64;
  const int i = (j < half) ? j : j - half;
  const float freq = __expf(-(float)i * (logf(10000.f) / 63.f));
  const float a = tv * freq;
  const float v = (j < half) ? sinf(a) : cosf(a);
  tg[b * 128 + j] = 0.5f * v * (1.0f + erff(v * 0.70710678118654752f));
}

// ---------------------------------------------------------------------------

struct CT  { const float *in_b,*in_w,*l11_b,*l11_w,*l12_b,*l12_w,
                         *ln12_b,*ln12_g,*ln13_b,*ln13_g,
                         *out_b,*out_w,*proj_b,*proj_w; };
struct MLP { const float *b,*beta,*g,*w; };

extern "C" void kernel_launch(void* const* d_in, const int* in_sizes, int n_in,
                              void* d_out, int out_size, void* d_ws, size_t ws_size,
                              hipStream_t stream)
{
  (void)in_sizes; (void)n_in; (void)out_size; (void)ws_size;
  const float* x = (const float*)d_in[0];   // (2,4,8192)
  const int*   t = (const int*)d_in[1];     // (2,)

  // ---- params in jax-pytree sorted-key order ----
  int pi = 2;
  auto nx = [&]() { return (const float*)d_in[pi++]; };
  auto ldMLP = [&]() { MLP m; m.b=nx(); m.beta=nx(); m.g=nx(); m.w=nx(); return m; };
  auto ldCT  = [&]() { CT c; c.in_b=nx(); c.in_w=nx(); c.l11_b=nx(); c.l11_w=nx();
                       c.l12_b=nx(); c.l12_w=nx(); c.ln12_b=nx(); c.ln12_g=nx();
                       c.ln13_b=nx(); c.ln13_g=nx(); c.out_b=nx(); c.out_w=nx();
                       c.proj_b=nx(); c.proj_w=nx(); return c; };
  MLP dec2_m1 = ldMLP(), dec2_m2 = ldMLP(), dec3_m1 = ldMLP(), dec3_m2 = ldMLP();
  CT  enc1 = ldCT(), enc2p = ldCT(), enc3p = ldCT();
  MLP fin1 = ldMLP();
  const float *fin2_b = nx(), *fin2_w = nx();
  MLP h0skip = ldMLP();
  const float *lift_beta = nx();
  const float *lift_c1_b = nx(), *lift_c1_w = nx();
  const float *lift_c2_b = nx(), *lift_c2_w = nx();
  const float *lift_g = nx();
  CT midp = ldCT();
  const float *out1_b = nx(), *out1_w = nx();
  const float *out2_b = nx(), *out2_w = nx();
  const float *tmlp_b[7], *tmlp_w[7];
  for (int j = 0; j < 7; ++j) { tmlp_b[j] = nx(); tmlp_w[j] = nx(); }

  // ---- workspace bump allocator ----
  size_t off = 0;
  char* base = (char*)d_ws;
  auto allocB = [&](size_t bytes) -> void* {
    size_t a = (off + 255) & ~(size_t)255; off = a + bytes; return (void*)(base + a);
  };
  auto alloc = [&](size_t elems) -> float* { return (float*)allocB(elems * 4); };

  // persistent buffers
  float* Sbuf = alloc((size_t)2048 * 8192);        // reused per (batch,head); lives in L2
  float* tg   = alloc(2 * 128);
  static const int tdims[7] = {128, 256, 512, 512, 256, 128, 64};
  float* tp[7]; for (int j = 0; j < 7; ++j) tp[j] = alloc((size_t)2 * tdims[j]);
  float* t0   = alloc((size_t)2 * 64 * 8192);
  float* h0   = alloc((size_t)2 * 64 * 8192);
  int*   idx1 = (int*)allocB((size_t)2 * 2048 * 4);
  float* xyz1 = alloc((size_t)2 * 3 * 2048);
  float* g1   = alloc((size_t)2 * 64 * 2048);
  float* h1   = alloc((size_t)2 * 128 * 2048);
  int*   idx2 = (int*)allocB((size_t)2 * 512 * 4);
  float* xyz2 = alloc((size_t)2 * 3 * 512);
  float* g2   = alloc((size_t)2 * 128 * 512);
  float* h2   = alloc((size_t)2 * 256 * 512);
  int*   idx3 = (int*)allocB((size_t)2 * 128 * 4);
  float* g3   = alloc((size_t)2 * 256 * 128);
  float* h3   = alloc((size_t)2 * 512 * 128);
  float* midb = alloc((size_t)2 * 512 * 128);
  float* upm  = alloc((size_t)2 * 512 * 512);
  float* cat3 = alloc((size_t)2 * 768 * 512);
  float* d3   = alloc((size_t)2 * 256 * 512);
  float* d3b  = alloc((size_t)2 * 256 * 512);
  float* upd3 = alloc((size_t)2 * 256 * 2048);
  float* cat2 = alloc((size_t)2 * 384 * 2048);
  float* d2   = alloc((size_t)2 * 128 * 2048);
  float* d2b  = alloc((size_t)2 * 128 * 2048);
  float* d2r  = alloc((size_t)2 * 128 * 4096);
  float* h0s  = alloc((size_t)2 * 64 * 4096);
  float* cat1 = alloc((size_t)2 * 192 * 4096);
  float* fint = alloc((size_t)2 * 64 * 4096);
  float* d1b  = alloc((size_t)2 * 64 * 4096);
  float* o1b  = alloc((size_t)2 * 32 * 4096);

  // ---- launch helpers ----
  // gemm: converts A (M x K, strides ars/acs) and B (K x N, strides brs/bcs)
  // into K-contiguous f16 mirrors, then runs the WMMA GEMM.
  auto gemm = [&](int M, int K, int N,
                  const float* A, long long ars, long long acs, long long abs_,
                  const float* Bm, long long brs, long long bcs, long long bbs,
                  const float* bias, const float* res, long long rbs,
                  float alpha, int act,
                  float* Y, long long yrs, long long ycs, long long ybs, int nb) {
    const size_t mark = off;
    _Float16* A16 = (_Float16*)allocB((size_t)nb * M * K * 2);
    _Float16* B16 = (_Float16*)allocB((size_t)nb * N * K * 2);
    {
      dim3 gr((unsigned)(((long long)M * K + 255) / 256), (unsigned)nb);
      k_cvt<<<gr, 256, 0, stream>>>(A, ars, acs, abs_, A16, M, K, (long long)M * K);
    }
    {
      // mirror rows are n: src row stride = bcs (per n), col stride = brs (per k)
      dim3 gr((unsigned)(((long long)N * K + 255) / 256), (unsigned)nb);
      k_cvt<<<gr, 256, 0, stream>>>(Bm, bcs, brs, bbs, B16, N, K, (long long)N * K);
    }
    dim3 g((unsigned)((N + 63) / 64), (unsigned)((M + 15) / 16), (unsigned)nb);
    k_gemm<<<g, 128, 0, stream>>>(M, K, N, A16, (long long)M * K,
                                  B16, (long long)N * K,
                                  bias, res, rbs, alpha, act, Y, yrs, ycs, ybs);
    off = mark;   // mirrors recycled next launch (stream-ordered, safe)
  };
  auto lnf = [&](const float* X, float* Y, const float* g, const float* bt, int C, int N) {
    dim3 gr((unsigned)((N + 255) / 256), 2);
    k_ln<<<gr, 256, 0, stream>>>(X, Y, g, bt, C, N, (long long)C * N);
  };
  auto gnf = [&](const float* X, float* Y, const float* g, const float* bt,
                 int C, int N, int act) {
    dim3 gr(4, 2);
    k_gn<<<gr, 256, 0, stream>>>(X, Y, g, bt, C, N, 4, act, (long long)C * N);
  };
  auto addvec = [&](float* Y, const float* V, int C, int N) {
    dim3 gr((unsigned)(((long long)C * N + 255) / 256), 2);
    k_addvec<<<gr, 256, 0, stream>>>(Y, V, C, (long long)N, (long long)C * N);
  };
  auto gather = [&](const float* src, long long sbs, int C, int Ns,
                    const int* idx, long long ibs, int M, float* dst, long long dbs) {
    dim3 gr((unsigned)(((long long)C * M + 255) / 256), 2);
    k_gather<<<gr, 256, 0, stream>>>(src, sbs, C, Ns, idx, ibs, M, dst, dbs);
  };
  auto up = [&](const float* src, long long sbs, int C, int Ns,
                float* dst, long long dbs, int Nt) {
    dim3 gr((unsigned)(((long long)C * Nt + 255) / 256), 2);
    k_up<<<gr, 256, 0, stream>>>(src, sbs, C, Ns, dst, dbs, Nt);
  };
  auto copych = [&](const float* src, long long sbs, int Cs, int N,
                    float* dst, long long dbs, int co) {
    dim3 gr((unsigned)(((long long)Cs * N + 255) / 256), 2);
    k_copych<<<gr, 256, 0, stream>>>(src, sbs, Cs, N, dst, dbs, co);
  };

  // cross-transformer block: src1 (di,Sq), src2 (di,Sk) -> out (dc,Sq)
  auto run_ct = [&](const CT& p, int di, int dc, int ff, int Sq, int Sk,
                    const float* src1, const float* src2, float* out) {
    const size_t mark = off;
    float* s1  = alloc((size_t)2 * dc * Sq);
    float* s2  = alloc((size_t)2 * dc * Sk);
    float* s1n = alloc((size_t)2 * dc * Sq);
    float* s2n = alloc((size_t)2 * dc * Sk);
    float* qh  = alloc((size_t)2 * dc * Sq);
    float* kh  = alloc((size_t)2 * dc * Sk);
    float* vh  = alloc((size_t)2 * dc * Sk);
    float* ob  = alloc((size_t)2 * dc * Sq);
    float* s1L = alloc((size_t)2 * dc * Sq);
    float* ffh = alloc((size_t)2 * ff * Sq);
    const long long bq = (long long)dc * Sq, bk = (long long)dc * Sk;

    gemm(dc, di, Sq, p.proj_w, di, 1, 0, src1, Sq, 1, (long long)di * Sq,
         p.proj_b, nullptr, 0, 1.f, 0, s1, Sq, 1, bq, 2);
    gemm(dc, di, Sk, p.proj_w, di, 1, 0, src2, Sk, 1, (long long)di * Sk,
         p.proj_b, nullptr, 0, 1.f, 0, s2, Sk, 1, bk, 2);
    lnf(s1, s1n, p.ln13_g, p.ln13_b, dc, Sq);
    lnf(s2, s2n, p.ln13_g, p.ln13_b, dc, Sk);
    gemm(dc, dc, Sq, p.in_w, dc, 1, 0, s1n, Sq, 1, bq,
         p.in_b, nullptr, 0, 1.f, 0, qh, Sq, 1, bq, 2);
    gemm(dc, dc, Sk, p.in_w + (long long)dc * dc, dc, 1, 0, s2n, Sk, 1, bk,
         p.in_b + dc, nullptr, 0, 1.f, 0, kh, Sk, 1, bk, 2);
    gemm(dc, dc, Sk, p.in_w + 2ll * dc * dc, dc, 1, 0, s2, Sk, 1, bk,
         p.in_b + 2 * dc, nullptr, 0, 1.f, 0, vh, Sk, 1, bk, 2);

    const int dh = dc / 4;
    const float scale = 1.0f / sqrtf((float)dh);
    for (int b = 0; b < 2; ++b)
      for (int h = 0; h < 4; ++h) {
        const float* q = qh + (long long)b * bq + (long long)h * dh * Sq;
        const float* k = kh + (long long)b * bk + (long long)h * dh * Sk;
        const float* v = vh + (long long)b * bk + (long long)h * dh * Sk;
        float* oh = ob + (long long)b * bq + (long long)h * dh * Sq;
        // S[q,k] = scale * qh^T kh   (A transposed via strides)
        gemm(Sq, dh, Sk, q, 1, Sq, 0, k, Sk, 1, 0,
             nullptr, nullptr, 0, scale, 0, Sbuf, Sk, 1, 0, 1);
        k_softmax<<<Sq, 256, 0, stream>>>(Sbuf, Sk);
        // o[d,q] = vh @ P^T
        gemm(dh, Sk, Sq, v, Sk, 1, 0, Sbuf, 1, Sk, 0,
             nullptr, nullptr, 0, 1.f, 0, oh, Sq, 1, 0, 1);
      }
    // out-proj + residual: s1 = s1 + Wout @ o + bout
    gemm(dc, dc, Sq, p.out_w, dc, 1, 0, ob, Sq, 1, bq,
         p.out_b, s1, bq, 1.f, 0, s1, Sq, 1, bq, 2);
    lnf(s1, s1L, p.ln12_g, p.ln12_b, dc, Sq);
    gemm(ff, dc, Sq, p.l11_w, dc, 1, 0, s1L, Sq, 1, bq,
         p.l11_b, nullptr, 0, 1.f, 1 /*gelu*/, ffh, Sq, 1, (long long)ff * Sq, 2);
    gemm(dc, ff, Sq, p.l12_w, ff, 1, 0, ffh, Sq, 1, (long long)ff * Sq,
         p.l12_b, s1L, bq, 1.f, 0, out, Sq, 1, bq, 2);
    off = mark;  // free ct temporaries (stream-ordered reuse is safe)
  };

  auto mlpb = [&](const MLP& p, int Ci, int Co, int N,
                  const float* X, long long xrs, long long xbs, float* Y) {
    const size_t mark = off;
    float* tb = alloc((size_t)2 * Co * N);
    gemm(Co, Ci, N, p.w, Ci, 1, 0, X, xrs, 1, xbs,
         p.b, nullptr, 0, 1.f, 0, tb, N, 1, (long long)Co * N, 2);
    gnf(tb, Y, p.g, p.beta, Co, N, 2 /*silu*/);
    off = mark;
  };

  // ======================= forward =======================
  k_temb<<<2, 128, 0, stream>>>(t, tg);
  for (int j = 0; j < 7; ++j)
    gemm(tdims[j], 128, 1, tmlp_w[j], 128, 1, 0, tg, 1, 1, 128,
         tmlp_b[j], nullptr, 0, 1.f, 0, tp[j], 1, 1, tdims[j], 2);

  // lift: conv(4->64) -> GN(gelu) -> conv(64->64)
  gemm(64, 4, 8192, lift_c1_w, 4, 1, 0, x, 8192, 1, 4ll * 8192,
       lift_c1_b, nullptr, 0, 1.f, 0, t0, 8192, 1, 64ll * 8192, 2);
  gnf(t0, t0, lift_g, lift_beta, 64, 8192, 1 /*gelu*/);
  gemm(64, 64, 8192, lift_c2_w, 64, 1, 0, t0, 8192, 1, 64ll * 8192,
       lift_c2_b, nullptr, 0, 1.f, 0, h0, 8192, 1, 64ll * 8192, 2);

  // encoder pyramid
  k_fps<<<2, 256, 0, stream>>>(x, 4ll * 8192, 8192, 2048, idx1, 2048);
  gather(h0, 64ll * 8192, 64, 8192, idx1, 2048, 2048, g1, 64ll * 2048);
  run_ct(enc1, 64, 128, 512, 2048, 8192, g1, h0, h1);
  addvec(h1, tp[0], 128, 2048);
  gather(x, 4ll * 8192, 3, 8192, idx1, 2048, 2048, xyz1, 3ll * 2048);

  k_fps<<<2, 256, 0, stream>>>(xyz1, 3ll * 2048, 2048, 512, idx2, 512);
  gather(h1, 128ll * 2048, 128, 2048, idx2, 512, 512, g2, 128ll * 512);
  run_ct(enc2p, 128, 256, 1024, 512, 2048, g2, h1, h2);
  addvec(h2, tp[1], 256, 512);
  gather(xyz1, 3ll * 2048, 3, 2048, idx2, 512, 512, xyz2, 3ll * 512);

  k_fps<<<2, 256, 0, stream>>>(xyz2, 3ll * 512, 512, 128, idx3, 128);
  gather(h2, 256ll * 512, 256, 512, idx3, 128, 128, g3, 256ll * 128);
  run_ct(enc3p, 256, 512, 2048, 128, 512, g3, h2, h3);
  addvec(h3, tp[2], 512, 128);

  run_ct(midp, 512, 512, 1024, 128, 128, h3, h3, midb);
  addvec(midb, tp[3], 512, 128);

  // decoder 3
  up(midb, 512ll * 128, 512, 128, upm, 512ll * 512, 512);
  copych(upm, 512ll * 512, 512, 512, cat3, 768ll * 512, 0);
  copych(h2, 256ll * 512, 256, 512, cat3, 768ll * 512, 512);
  mlpb(dec3_m1, 768, 256, 512, cat3, 512, 768ll * 512, d3);
  addvec(d3, tp[4], 256, 512);
  mlpb(dec3_m2, 256, 256, 512, d3, 512, 256ll * 512, d3b);

  // decoder 2
  up(d3b, 256ll * 512, 256, 512, upd3, 256ll * 2048, 2048);
  copych(upd3, 256ll * 2048, 256, 2048, cat2, 384ll * 2048, 0);
  copych(h1, 128ll * 2048, 128, 2048, cat2, 384ll * 2048, 256);
  mlpb(dec2_m1, 384, 128, 2048, cat2, 2048, 384ll * 2048, d2);
  addvec(d2, tp[5], 128, 2048);
  mlpb(dec2_m2, 128, 128, 2048, d2, 2048, 128ll * 2048, d2b);
  up(d2b, 128ll * 2048, 128, 2048, d2r, 128ll * 4096, 4096);

  // skip from h0 (columns 4096..8191, strided view)
  mlpb(h0skip, 64, 64, 4096, h0 + 4096, 8192, 64ll * 8192, h0s);
  copych(d2r, 128ll * 4096, 128, 4096, cat1, 192ll * 4096, 0);
  copych(h0s, 64ll * 4096, 64, 4096, cat1, 192ll * 4096, 128);

  // final head
  gemm(64, 192, 4096, fin1.w, 192, 1, 0, cat1, 4096, 1, 192ll * 4096,
       fin1.b, nullptr, 0, 1.f, 0, fint, 4096, 1, 64ll * 4096, 2);
  gnf(fint, fint, fin1.g, fin1.beta, 64, 4096, 1 /*gelu*/);
  gemm(64, 64, 4096, fin2_w, 64, 1, 0, fint, 4096, 1, 64ll * 4096,
       fin2_b, nullptr, 0, 1.f, 0, d1b, 4096, 1, 64ll * 4096, 2);
  addvec(d1b, tp[6], 64, 4096);
  gemm(32, 64, 4096, out1_w, 64, 1, 0, d1b, 4096, 1, 64ll * 4096,
       out1_b, nullptr, 0, 1.f, 1 /*gelu*/, o1b, 4096, 1, 32ll * 4096, 2);
  gemm(3, 32, 4096, out2_w, 32, 1, 0, o1b, 4096, 1, 32ll * 4096,
       out2_b, nullptr, 0, 1.f, 0, (float*)d_out, 4096, 1, 3ll * 4096, 2);
}
